// NeRFTrainer_12111807774817
// MI455X (gfx1250) — matrix operand order repaired and checked by reading, not compiled, not run on hardware
//
#include <hip/hip_runtime.h>
#include <hip/hip_bf16.h>
#include <math.h>

// ---------------------------------------------------------------------------
// CDNA5 / gfx1250 NeRF forward (coarse + fine) built around v_wmma_f32_16x16x32_f16.
// wave32, WGP. GEMM block tile 128x64, each wave owns a 32x32 output quad
// (4 WMMA accumulators). Double-buffered LDS staging via
// GLOBAL_LOAD_ASYNC_TO_LDS_B128 (ASYNCcnt, in-order completion: wait<=3 keeps
// the next k-block's loads in flight behind the WMMAs).
// ---------------------------------------------------------------------------

typedef __attribute__((ext_vector_type(16))) _Float16 v16h;
typedef __attribute__((ext_vector_type(8)))  _Float16 v8h;
typedef __attribute__((ext_vector_type(8)))  float    v8f;
typedef __attribute__((ext_vector_type(4)))  int      v4i;

#define BM 128
#define BN 64
#define KB 32

#if defined(__HIP_DEVICE_COMPILE__) && __has_builtin(__builtin_amdgcn_global_load_async_to_lds_b128)
#define NERF_HAVE_ASYNC_LDS 1
typedef __attribute__((address_space(1))) v4i as1_v4i;   // global int4
typedef __attribute__((address_space(3))) v4i as3_v4i;   // LDS int4
#endif

__device__ __forceinline__ void nerf_wait_async(int n)
{
#if defined(NERF_HAVE_ASYNC_LDS)
#if __has_builtin(__builtin_amdgcn_s_wait_asynccnt)
    if (n == 0) __builtin_amdgcn_s_wait_asynccnt(0);
    else        __builtin_amdgcn_s_wait_asynccnt(3);
#else
    if (n == 0) asm volatile("s_wait_asynccnt 0" ::: "memory");
    else        asm volatile("s_wait_asynccnt 3" ::: "memory");
#endif
#endif
    (void)n;
}

// ---------------------------------------------------------------------------
// Weight prep: W[K_real x N_real] f32 -> Wt[N_pad][K_pad] f16 (transposed, zero pad)
// ---------------------------------------------------------------------------
__global__ __launch_bounds__(256) void nerf_prep_w(
    const float* __restrict__ W, int K_real, int N_real,
    _Float16* __restrict__ Wt, int K_pad, int N_pad)
{
    int idx = blockIdx.x * 256 + threadIdx.x;
    int total = K_pad * N_pad;
    if (idx >= total) return;
    int n = idx / K_pad;
    int k = idx - n * K_pad;
    float v = (k < K_real && n < N_real) ? W[(size_t)k * N_real + n] : 0.0f;
    Wt[(size_t)n * K_pad + k] = (_Float16)v;
}

// ---------------------------------------------------------------------------
// Positional / direction encoding. Writes xenc (stride 64, col 63 zero) and
// direction encoding into Ad columns 256..287 (cols 283..287 zero), so the
// concat(h, d_enc) input of the Wd layer is materialized in place.
// ---------------------------------------------------------------------------
__global__ __launch_bounds__(256) void nerf_encode(
    const float* __restrict__ o, const float* __restrict__ dvec,
    const float* __restrict__ t, int NT, int ray0,
    _Float16* __restrict__ xenc, _Float16* __restrict__ Ad, int npts)
{
    int p = blockIdx.x * 256 + threadIdx.x;
    if (p >= npts) return;
    int ray = ray0 + p / NT;
    int s   = p - (p / NT) * NT;
    float tv = t[(size_t)ray * NT + s];

    float pos[3], dir[3];
#pragma unroll
    for (int c = 0; c < 3; ++c) {
        dir[c] = dvec[ray * 3 + c];
        pos[c] = o[ray * 3 + c] + dir[c] * tv;
    }

    _Float16* xr = xenc + (size_t)p * 64;
#pragma unroll
    for (int c = 0; c < 3; ++c) xr[c] = (_Float16)pos[c];
#pragma unroll
    for (int i = 0; i < 10; ++i) {
        float f = (float)(1 << i);
#pragma unroll
        for (int c = 0; c < 3; ++c) {
            float a = pos[c] * f;
            xr[3 + 6 * i + c]     = (_Float16)__sinf(a);
            xr[3 + 6 * i + 3 + c] = (_Float16)__cosf(a);
        }
    }
    xr[63] = (_Float16)0.0f;

    _Float16* dr = Ad + (size_t)p * 288 + 256;
#pragma unroll
    for (int c = 0; c < 3; ++c) dr[c] = (_Float16)dir[c];
#pragma unroll
    for (int i = 0; i < 4; ++i) {
        float f = (float)(1 << i);
#pragma unroll
        for (int c = 0; c < 3; ++c) {
            float a = dir[c] * f;
            dr[3 + 6 * i + c]     = (_Float16)__sinf(a);
            dr[3 + 6 * i + 3 + c] = (_Float16)__cosf(a);
        }
    }
#pragma unroll
    for (int k = 27; k < 32; ++k) dr[k] = (_Float16)0.0f;
}

// ---------------------------------------------------------------------------
// WMMA GEMM: out = act(A @ W + b).
//   A : M x lda (f16), uses K columns (K mult of 32, M mult of 128)
//   Wt: [N_pad][K_pad] f16 (transposed weights)
//   relu_f16 = 1 -> ReLU, f16 out ; 0 -> passthrough, f32 out
// Block: 256 threads (8 waves). Tile 128x64; wave (w&3) -> 32-row group,
// (w>>2) -> 32-col group; 4 accumulators per wave (32x32 output quad).
// Double-buffered LDS k-block pipeline.
// ---------------------------------------------------------------------------
__global__ __launch_bounds__(256) void nerf_gemm_wmma(
    const _Float16* __restrict__ A, int lda,
    const _Float16* __restrict__ Wt, int K_pad,
    const float* __restrict__ bias, int N_real,
    void* __restrict__ out, int ldo, int N_store, int relu_f16,
    int M, int K)
{
    __shared__ __align__(16) _Float16 sA[2][BM][KB];   // 2 x 8KB
    __shared__ __align__(16) _Float16 sW[2][BN][KB];   // 2 x 4KB

    const int tid  = threadIdx.x;
    const int wave = tid >> 5;
    const int lane = tid & 31;
    const int m0 = blockIdx.x * BM;
    const int n0 = blockIdx.y * BN;

    const int wm = (wave & 3) * 32;   // wave's 32-row group within tile
    const int wn = (wave >> 2) * 32;  // wave's 32-col group within tile

    // A fragment indices (16-bit A 16x32 layout, wave32)
    const int a_row0 = wm + (lane & 15);
    const int a_row1 = a_row0 + 16;
    const int a_kb   = (lane >> 4) * 8;
    // B fragment indices (16-bit B 32x16 layout, wave32)
    const int b_n0 = wn + (lane & 15);
    const int b_n1 = b_n0 + 16;
    const int b_kb = (lane >> 4) * 8;

    // Staging: 768 x 16B chunks per k-block (512 for sA, 256 for sW); 3/thread.
    const _Float16* gbase[3];
    _Float16*       lbase[3];
    int             lstep[3];   // halves between parity buffers
#pragma unroll
    for (int j = 0; j < 3; ++j) {
        int chunk = j * 256 + tid;
        if (chunk < 512) {
            int r = chunk >> 2;
            int c = (chunk & 3) * 8;
            gbase[j] = A + (size_t)(m0 + r) * lda + c;
            lbase[j] = &sA[0][r][c];
            lstep[j] = BM * KB;
        } else {
            int idx = chunk - 512;
            int r = idx >> 2;
            int c = (idx & 3) * 8;
            gbase[j] = Wt + (size_t)(n0 + r) * K_pad + c;
            lbase[j] = &sW[0][r][c];
            lstep[j] = BN * KB;
        }
    }

    auto issue = [&](int pp, int kb) {
#pragma unroll
        for (int j = 0; j < 3; ++j) {
            const _Float16* g = gbase[j] + kb;
            _Float16* l = lbase[j] + pp * lstep[j];
#if defined(NERF_HAVE_ASYNC_LDS)
            __builtin_amdgcn_global_load_async_to_lds_b128(
                (as1_v4i*)(void*)g, (as3_v4i*)(void*)l, 0, 0);
#else
            *(v8h*)l = *(const v8h*)g;
#endif
            // speculative prefetch two k-blocks ahead (global_prefetch_b8)
            __builtin_prefetch(g + 2 * KB, 0, 1);
        }
    };

    v8f acc00 = {};
    v8f acc01 = {};
    v8f acc10 = {};
    v8f acc11 = {};

    issue(0, 0);
    int p = 0;
    for (int kb = 0; kb < K; kb += KB) {
        const bool has_next = (kb + KB) < K;
        if (has_next) {
            issue(p ^ 1, kb + KB);   // overlap next k-block with this compute
            nerf_wait_async(3);      // older 3 loads (buffer p) have landed
        } else {
            nerf_wait_async(0);
        }
        __syncthreads();             // buffer p visible to all waves

        v8h a0lo = *(const v8h*)(&sA[p][a_row0][a_kb]);
        v8h a0hi = *(const v8h*)(&sA[p][a_row0][16 + a_kb]);
        v16h afrag0 = __builtin_shufflevector(a0lo, a0hi,
            0, 1, 2, 3, 4, 5, 6, 7, 8, 9, 10, 11, 12, 13, 14, 15);
        v8h a1lo = *(const v8h*)(&sA[p][a_row1][a_kb]);
        v8h a1hi = *(const v8h*)(&sA[p][a_row1][16 + a_kb]);
        v16h afrag1 = __builtin_shufflevector(a1lo, a1hi,
            0, 1, 2, 3, 4, 5, 6, 7, 8, 9, 10, 11, 12, 13, 14, 15);

        v8h b0lo = *(const v8h*)(&sW[p][b_n0][b_kb]);
        v8h b0hi = *(const v8h*)(&sW[p][b_n0][b_kb + 16]);
        v16h bfrag0 = __builtin_shufflevector(b0lo, b0hi,
            0, 1, 2, 3, 4, 5, 6, 7, 8, 9, 10, 11, 12, 13, 14, 15);
        v8h b1lo = *(const v8h*)(&sW[p][b_n1][b_kb]);
        v8h b1hi = *(const v8h*)(&sW[p][b_n1][b_kb + 16]);
        v16h bfrag1 = __builtin_shufflevector(b1lo, b1hi,
            0, 1, 2, 3, 4, 5, 6, 7, 8, 9, 10, 11, 12, 13, 14, 15);

        acc00 = __builtin_amdgcn_wmma_f32_16x16x32_f16(
            false, afrag0, false, bfrag0, (short)0, acc00, false, false);
        acc01 = __builtin_amdgcn_wmma_f32_16x16x32_f16(
            false, afrag0, false, bfrag1, (short)0, acc01, false, false);
        acc10 = __builtin_amdgcn_wmma_f32_16x16x32_f16(
            false, afrag1, false, bfrag0, (short)0, acc10, false, false);
        acc11 = __builtin_amdgcn_wmma_f32_16x16x32_f16(
            false, afrag1, false, bfrag1, (short)0, acc11, false, false);

        __syncthreads();             // all waves done reading buffer p (WAR)
        p ^= 1;
    }

    // Epilogue. C/D layout: VGPR r -> M = r + 8*(lane>=16), N = lane&15.
    const int rbase0 = m0 + wm + (lane >> 4) * 8;
    const int rbase1 = rbase0 + 16;
    const int ncol0  = n0 + wn + (lane & 15);
    const int ncol1  = ncol0 + 16;
    const float bia0 = (ncol0 < N_real) ? bias[ncol0] : 0.0f;
    const float bia1 = (ncol1 < N_real) ? bias[ncol1] : 0.0f;

#pragma unroll
    for (int r = 0; r < 8; ++r) {
        float v00 = acc00[r] + bia0;
        float v01 = acc01[r] + bia1;
        float v10 = acc10[r] + bia0;
        float v11 = acc11[r] + bia1;
        int row0 = rbase0 + r;
        int row1 = rbase1 + r;
        if (relu_f16) {
            v00 = v00 > 0.0f ? v00 : 0.0f;
            v01 = v01 > 0.0f ? v01 : 0.0f;
            v10 = v10 > 0.0f ? v10 : 0.0f;
            v11 = v11 > 0.0f ? v11 : 0.0f;
            _Float16* po0 = (_Float16*)out + (size_t)row0 * ldo;
            _Float16* po1 = (_Float16*)out + (size_t)row1 * ldo;
            if (ncol0 < N_store) { po0[ncol0] = (_Float16)v00; po1[ncol0] = (_Float16)v10; }
            if (ncol1 < N_store) { po0[ncol1] = (_Float16)v01; po1[ncol1] = (_Float16)v11; }
        } else {
            float* po0 = (float*)out + (size_t)row0 * ldo;
            float* po1 = (float*)out + (size_t)row1 * ldo;
            if (ncol0 < N_store) { po0[ncol0] = v00; po1[ncol0] = v10; }
            if (ncol1 < N_store) { po0[ncol1] = v01; po1[ncol1] = v11; }
        }
    }
}

// ---------------------------------------------------------------------------
// Volume rendering: one thread per ray (chunk-local), serial over samples.
// sig/rgbraw: per-point 16-wide f32 tiles (col 0 = sigma; cols 0..2 = rgb raw).
// ---------------------------------------------------------------------------
__global__ __launch_bounds__(256) void nerf_render(
    const float* __restrict__ sig, const float* __restrict__ rgbraw,
    const float* __restrict__ t, int NT, int ray0, int nrays,
    float* __restrict__ out_rgb, float* __restrict__ out_depth,
    float* __restrict__ out_w)
{
    int r = blockIdx.x * 256 + threadIdx.x;
    if (r >= nrays) return;
    int ray = ray0 + r;
    const float* trow = t + (size_t)ray * NT;

    float T = 1.0f, depth = 0.0f;
    float rgb[3] = {0.0f, 0.0f, 0.0f};
    for (int s = 0; s < NT; ++s) {
        size_t p = (size_t)r * NT + s;
        float sigma = sig[p * 16];
        sigma = sigma > 0.0f ? sigma : 0.0f;
        float delta = (s + 1 < NT) ? (trow[s + 1] - trow[s]) : 1e10f;
        float alpha = 1.0f - __expf(-sigma * delta);
        float w = alpha * T;
        T *= (1.0f - alpha + 1e-10f);
        out_w[(size_t)ray * NT + s] = w;
#pragma unroll
        for (int c = 0; c < 3; ++c) {
            float raw = rgbraw[p * 16 + c];
            rgb[c] += w * (1.0f / (1.0f + __expf(-raw)));
        }
        depth += w * trow[s];
    }
#pragma unroll
    for (int c = 0; c < 3; ++c) out_rgb[ray * 3 + c] = rgb[c];
    out_depth[ray] = depth;
}

// ---------------------------------------------------------------------------
// Hierarchical sampling: per-ray inverse-CDF sampling (hash RNG stands in for
// the JAX PRNG stream), then block-per-ray rank sort merges t_vals + t_fine.
// ---------------------------------------------------------------------------
__device__ inline float nerf_rng01(unsigned x)
{
    x = (x ^ 61u) ^ (x >> 16);
    x *= 9u;
    x ^= x >> 4;
    x *= 0x27d4eb2du;
    x ^= x >> 15;
    return (float)(x >> 8) * (1.0f / 16777216.0f);
}

__global__ __launch_bounds__(256) void nerf_sample_pdf(
    const float* __restrict__ tvals,   // R x 64
    const float* __restrict__ wcoarse, // R x 64
    float* __restrict__ tfine)         // R x 128
{
    int ray = blockIdx.x * 256 + threadIdx.x;
    if (ray >= 1024) return;

    const float* tr = tvals + (size_t)ray * 64;
    const float* wr = wcoarse + (size_t)ray * 64;

    float bins[63];
    float cdf[63];
    for (int i = 0; i < 63; ++i) bins[i] = 0.5f * (tr[i] + tr[i + 1]);

    float sum = 0.0f;
    for (int i = 0; i < 62; ++i) sum += wr[i + 1] + 1e-5f;
    float inv = 1.0f / sum;
    cdf[0] = 0.0f;
    float acc = 0.0f;
    for (int i = 0; i < 62; ++i) {
        acc += (wr[i + 1] + 1e-5f) * inv;
        cdf[i + 1] = acc;
    }

    for (int j = 0; j < 128; ++j) {
        float u = nerf_rng01((unsigned)(ray * 131 + j) * 2654435761u + 12345u);
        int ind = 0;
        for (int i = 0; i < 63; ++i) ind += (cdf[i] <= u) ? 1 : 0;
        int below = ind - 1; below = below < 0 ? 0 : (below > 62 ? 62 : below);
        int above = ind > 62 ? 62 : ind;
        float clo = cdf[below], chi = cdf[above];
        float blo = bins[below], bhi = bins[above];
        float den = chi - clo;
        if (den < 1e-5f) den = 1.0f;
        float frac = (u - clo) / den;
        tfine[(size_t)ray * 128 + j] = blo + frac * (bhi - blo);
    }
}

__global__ __launch_bounds__(192) void nerf_merge_sort(
    const float* __restrict__ tvals,  // R x 64 (sorted)
    const float* __restrict__ tfine,  // R x 128
    float* __restrict__ tall)         // R x 192 sorted
{
    __shared__ float v[192];
    int ray = blockIdx.x;
    int i = threadIdx.x;
    v[i] = (i < 64) ? tvals[(size_t)ray * 64 + i]
                    : tfine[(size_t)ray * 128 + (i - 64)];
    __syncthreads();
    float x = v[i];
    int rank = 0;
    for (int j = 0; j < 192; ++j) {
        float y = v[j];
        rank += (y < x || (y == x && j < i)) ? 1 : 0;
    }
    tall[(size_t)ray * 192 + rank] = x;
}

// ---------------------------------------------------------------------------
// Host orchestration.
// Inputs: 0 origins, 1 dirs, 2 t_vals, 3..12 coarse params, 13..22 fine params
// (each param dict flattens as W0,b0,W1,b1,Ws,bs,Wd,bd,Wr,br).
// Output: rgb_c | rgb_f | depth_c | depth_f | weights_c | weights_f.
// ---------------------------------------------------------------------------
extern "C" void kernel_launch(void* const* d_in, const int* in_sizes, int n_in,
                              void* d_out, int out_size, void* d_ws, size_t ws_size,
                              hipStream_t stream)
{
    (void)in_sizes; (void)n_in; (void)out_size; (void)ws_size;

    const float* o    = (const float*)d_in[0];
    const float* dvec = (const float*)d_in[1];
    const float* tv   = (const float*)d_in[2];
    const float* pc[10];
    const float* pf[10];
    for (int i = 0; i < 10; ++i) {
        pc[i] = (const float*)d_in[3 + i];
        pf[i] = (const float*)d_in[13 + i];
    }

    char* ws = (char*)d_ws;
    size_t cur = 0;
    auto alloc = [&](size_t bytes) -> char* {
        char* p = ws + cur;
        cur = (cur + bytes + 255) & ~(size_t)255;
        return p;
    };

    struct LayerW { int K_real, N_real, K_pad, N_pad; };
    const LayerW lw[5] = {
        { 63, 256,  64, 256},   // W0
        {256, 256, 256, 256},   // W1
        {256,   1, 256,  64},   // Ws
        {283, 128, 288, 128},   // Wd
        {128,   3, 128,  64},   // Wr
    };
    _Float16* wtc[5];
    _Float16* wtf[5];
    for (int l = 0; l < 5; ++l) {
        size_t b = (size_t)lw[l].K_pad * lw[l].N_pad * sizeof(_Float16);
        wtc[l] = (_Float16*)alloc(b);
        wtf[l] = (_Float16*)alloc(b);
    }

    const int P = 65536;  // max chunk point count (coarse full pass)
    _Float16* xenc   = (_Float16*)alloc((size_t)P * 64  * sizeof(_Float16));
    _Float16* h0     = (_Float16*)alloc((size_t)P * 256 * sizeof(_Float16));
    _Float16* Ad     = (_Float16*)alloc((size_t)P * 288 * sizeof(_Float16));
    _Float16* hd     = (_Float16*)alloc((size_t)P * 128 * sizeof(_Float16));
    float*    sig    = (float*)alloc((size_t)P * 16 * sizeof(float));
    float*    rgbraw = (float*)alloc((size_t)P * 16 * sizeof(float));
    float*    tfine  = (float*)alloc((size_t)1024 * 128 * sizeof(float));
    float*    tall   = (float*)alloc((size_t)1024 * 192 * sizeof(float));

    // weight prep (both param sets)
    for (int l = 0; l < 5; ++l) {
        int tot = lw[l].K_pad * lw[l].N_pad;
        int g = (tot + 255) / 256;
        nerf_prep_w<<<g, 256, 0, stream>>>(pc[2 * l], lw[l].K_real, lw[l].N_real,
                                           wtc[l], lw[l].K_pad, lw[l].N_pad);
        nerf_prep_w<<<g, 256, 0, stream>>>(pf[2 * l], lw[l].K_real, lw[l].N_real,
                                           wtf[l], lw[l].K_pad, lw[l].N_pad);
    }

    float* out = (float*)d_out;
    float* rgb_c = out;
    float* rgb_f = out + 3072;
    float* dep_c = out + 6144;
    float* dep_f = out + 7168;
    float* w_c   = out + 8192;
    float* w_f   = out + 73728;

    auto run_mlp = [&](_Float16* const* wt, const float* const* prm, int M) {
        dim3 blk(256);
        nerf_gemm_wmma<<<dim3(M / BM, 4), blk, 0, stream>>>(
            xenc, 64, wt[0], 64, prm[1], 256, (void*)h0, 256, 256, 1, M, 64);
        nerf_gemm_wmma<<<dim3(M / BM, 4), blk, 0, stream>>>(
            h0, 256, wt[1], 256, prm[3], 256, (void*)Ad, 288, 256, 1, M, 256);
        nerf_gemm_wmma<<<dim3(M / BM, 1), blk, 0, stream>>>(
            Ad, 288, wt[2], 256, prm[5], 1, (void*)sig, 16, 16, 0, M, 256);
        nerf_gemm_wmma<<<dim3(M / BM, 2), blk, 0, stream>>>(
            Ad, 288, wt[3], 288, prm[7], 128, (void*)hd, 128, 128, 1, M, 288);
        nerf_gemm_wmma<<<dim3(M / BM, 1), blk, 0, stream>>>(
            hd, 128, wt[4], 128, prm[9], 3, (void*)rgbraw, 16, 16, 0, M, 128);
    };

    // ---- coarse pass: 1024 rays x 64 samples ----
    nerf_encode<<<(65536 + 255) / 256, 256, 0, stream>>>(
        o, dvec, tv, 64, 0, xenc, Ad, 65536);
    run_mlp(wtc, pc, 65536);
    nerf_render<<<4, 256, 0, stream>>>(sig, rgbraw, tv, 64, 0, 1024,
                                       rgb_c, dep_c, w_c);

    // ---- hierarchical resampling ----
    nerf_sample_pdf<<<4, 256, 0, stream>>>(tv, w_c, tfine);
    nerf_merge_sort<<<1024, 192, 0, stream>>>(tv, tfine, tall);

    // ---- fine pass: 4 chunks of 256 rays x 192 samples ----
    for (int c = 0; c < 4; ++c) {
        int ray0 = c * 256;
        int npts = 256 * 192;  // 49152, multiple of 128
        nerf_encode<<<(npts + 255) / 256, 256, 0, stream>>>(
            o, dvec, tall, 192, ray0, xenc, Ad, npts);
        run_mlp(wtf, pf, npts);
        nerf_render<<<1, 256, 0, stream>>>(sig, rgbraw, tall, 192, ray0, 256,
                                           rgb_f, dep_f, w_f);
    }
}
